// PairwiseUpdateBlock_89386859364540
// MI455X (gfx1250) — compile-verified, matrix operand order checked by reading, not run on hardware
//
#include <hip/hip_runtime.h>
#include <hip/hip_bf16.h>
#include <math.h>

#define BDIM 2
#define LDIM 768
#define EDIM 1024
#define CDIM 64
#define LOWD 64
#define NGRP 32
#define EPSV 1e-5f

#define BLKS_PER_CH ((LDIM * LDIM) / 256)   // 2304 partial-sum blocks per (b,c)

typedef __attribute__((ext_vector_type(16))) __bf16 v16bf;
typedef __attribute__((ext_vector_type(8)))  float  v8f;

// K index for the CDNA5 16-bit A-matrix (16x32) / B^T layout:
// lanes 0-15: M=lane, VGPR0..3 hold K={0..7}, VGPR4..7 hold K={16..23}
// lanes 16-31: same M, K offset +8.   e = element index 0..15 in v16bf.
__device__ __forceinline__ int wk16(int lane, int e) {
  int r = e >> 1;
  return ((r < 4) ? (2 * r) : (16 + 2 * (r - 4))) + (e & 1) + ((lane & 16) ? 8 : 0);
}

// ---------------------------------------------------------------- K1: a = x @ W_sl^T + b_sl
__global__ __launch_bounds__(256) void k1_seq_proj(const float* __restrict__ x,
                                                   const float* __restrict__ Wsl,
                                                   const float* __restrict__ bsl,
                                                   float* __restrict__ a) {
  int idx = blockIdx.x * 256 + threadIdx.x;          // over B*L*LOW
  if (idx >= BDIM * LDIM * LOWD) return;
  int d = idx & 63;
  int l = (idx >> 6) % LDIM;
  int b = idx / (LDIM * LOWD);
  const float* xr = x + (size_t)(b * LDIM + l) * EDIM;
  const float* wr = Wsl + (size_t)d * EDIM;
  float s = 0.f;
#pragma unroll 8
  for (int e = 0; e < EDIM; ++e) s += xr[e] * wr[e];
  a[idx] = s + bsl[d];
}

// ---------------------------------------------------------------- K2: outer-projection GEMM
// (WMMA bf16) + b_op + LayerNorm(C) + residual with x_pairwise -> xp (stored in d_out)
__global__ __launch_bounds__(256) void k2_pair(const float* __restrict__ xpw,
                                               const float* __restrict__ a,
                                               const float* __restrict__ Wop,
                                               const float* __restrict__ bop,
                                               const float* __restrict__ lnw,
                                               const float* __restrict__ lnb,
                                               float* __restrict__ xp) {
  __shared__ float sAi[16][LOWD];
  __shared__ float sAj[16][LOWD];
  __shared__ float sW[3 * CDIM];          // [0..63]=b_op  [64..127]=ln_w  [128..191]=ln_b
  __shared__ float sTile[8][16][68];      // per-wave j x c tile (padded)
  __shared__ float sStat[8][16][2];       // per-wave row mean/inv

  const int tid  = threadIdx.x;
  const int lane = tid & 31;
  const int w    = tid >> 5;
  const int jt = blockIdx.x, it = blockIdx.y, b = blockIdx.z;

  for (int t = tid; t < 16 * LOWD; t += 256) {
    int r = t >> 6, d = t & 63;
    sAi[r][d] = a[((size_t)b * LDIM + it * 16 + r) * LOWD + d];
    sAj[r][d] = a[((size_t)b * LDIM + jt * 16 + r) * LOWD + d];
  }
  for (int t = tid; t < 3 * CDIM; t += 256) {
    int which = t >> 6, c = t & 63;
    sW[t] = (which == 0) ? bop[c] : (which == 1 ? lnw[c] : lnb[c]);
  }
  __syncthreads();

  // B = W_op^T  (B[k=d][n=c]), hoisted: same for every tile
  v16bf Bt[4][2];
#pragma unroll
  for (int ct = 0; ct < 4; ++ct)
#pragma unroll
    for (int kh = 0; kh < 2; ++kh)
#pragma unroll
      for (int e = 0; e < 16; ++e) {
        int n = lane & 15;
        int d = wk16(lane, e) + 32 * kh;
        Bt[ct][kh][e] = (__bf16)Wop[(ct * 16 + n) * LOWD + d];
      }

  for (int s = 0; s < 2; ++s) {
    const int il = w * 2 + s;               // local i row handled by this wave
    const int i  = it * 16 + il;

    // A[m=j_local][k=d] = a[j,d] * a[i,d]  (bf16)
    v16bf A[2];
#pragma unroll
    for (int kh = 0; kh < 2; ++kh)
#pragma unroll
      for (int e = 0; e < 16; ++e) {
        int m = lane & 15;
        int d = wk16(lane, e) + 32 * kh;
        A[kh][e] = (__bf16)(sAj[m][d] * sAi[il][d]);
      }

    v8f acc[4];
#pragma unroll
    for (int ct = 0; ct < 4; ++ct) {
      v8f z{};
      z = __builtin_amdgcn_wmma_f32_16x16x32_bf16(false, A[0], false, Bt[ct][0],
                                                  (short)0, z, false, false);
      z = __builtin_amdgcn_wmma_f32_16x16x32_bf16(false, A[1], false, Bt[ct][1],
                                                  (short)0, z, false, false);
      acc[ct] = z;
    }

    // D layout: M(=j_local) = v + 8*(lane>=16), N(=c_local) = lane&15
#pragma unroll
    for (int ct = 0; ct < 4; ++ct)
#pragma unroll
      for (int v = 0; v < 8; ++v) {
        int jl = v + ((lane >> 4) << 3);
        int c  = ct * 16 + (lane & 15);
        sTile[w][jl][c] = acc[ct][v] + sW[c];
      }

    if (lane < 16) {                        // per-(i,j) stats over the 64 channels
      float sm = 0.f, sq = 0.f;
#pragma unroll 8
      for (int c = 0; c < 64; ++c) { float v = sTile[w][lane][c]; sm += v; sq += v * v; }
      float mean = sm * (1.f / 64.f);
      float var  = sq * (1.f / 64.f) - mean * mean;
      sStat[w][lane][0] = mean;
      sStat[w][lane][1] = rsqrtf(var + EPSV);
    }

#pragma unroll 4
    for (int t = 0; t < 32; ++t) {          // LN + residual, coalesced over j
      int c  = 2 * t + (lane >> 4);
      int jl = lane & 15;
      float mean = sStat[w][jl][0], inv = sStat[w][jl][1];
      float v = (sTile[w][jl][c] - mean) * inv * sW[64 + c] + sW[128 + c];
      size_t idx = (((size_t)b * CDIM + c) * LDIM + i) * LDIM + (size_t)(jt * 16 + jl);
      xp[idx] = xpw[idx] + v;
    }
  }
}

// ---------------------------------------------------------------- K3: depthwise 3x3 + deterministic GN partials
__global__ __launch_bounds__(256) void k3_dwconv(const float* __restrict__ xp,
                                                 const float* __restrict__ dww,
                                                 float* __restrict__ h1,
                                                 float* __restrict__ psum,
                                                 float* __restrict__ psq) {
  size_t idx = (size_t)blockIdx.x * 256 + threadIdx.x;   // over B*C*L*L
  int j = (int)(idx % LDIM);
  int i = (int)((idx / LDIM) % LDIM);
  int c = (int)((idx / ((size_t)LDIM * LDIM)) % CDIM);
  int b = (int)(idx / ((size_t)LDIM * LDIM * CDIM));
  const float* base = xp + ((size_t)b * CDIM + c) * LDIM * LDIM;
  const float* wt = dww + c * 9;

  // pull the next stencil row toward the caches (global_prefetch_b8)
  if (i + 2 < LDIM) __builtin_prefetch(base + (size_t)(i + 2) * LDIM + j, 0, 3);

  float s = 0.f;
#pragma unroll
  for (int ky = 0; ky < 3; ++ky) {
    int ii = i + ky - 1;
    if (ii < 0 || ii >= LDIM) continue;
#pragma unroll
    for (int kx = 0; kx < 3; ++kx) {
      int jj = j + kx - 1;
      if (jj < 0 || jj >= LDIM) continue;
      s += wt[ky * 3 + kx] * base[(size_t)ii * LDIM + jj];
    }
  }
  h1[idx] = s;

  __shared__ float rs[256], rq[256];
  rs[threadIdx.x] = s; rq[threadIdx.x] = s * s;
  __syncthreads();
  for (int o = 128; o > 0; o >>= 1) {
    if (threadIdx.x < o) { rs[threadIdx.x] += rs[threadIdx.x + o]; rq[threadIdx.x] += rq[threadIdx.x + o]; }
    __syncthreads();
  }
  if (threadIdx.x == 0) {                   // deterministic per-block partials
    int bc     = blockIdx.x / BLKS_PER_CH;  // (b*C + c), block fully inside one channel
    int within = blockIdx.x % BLKS_PER_CH;
    psum[(size_t)bc * BLKS_PER_CH + within] = rs[0];
    psq [(size_t)bc * BLKS_PER_CH + within] = rq[0];
  }
}

// ---------------------------------------------------------------- K3b: per-(b,c) fixed-tree reduction of partials
__global__ __launch_bounds__(256) void k3b_reduce(const float* __restrict__ psum,
                                                  const float* __restrict__ psq,
                                                  float* __restrict__ csum,
                                                  float* __restrict__ csq) {
  const int bc = blockIdx.x;                // 0 .. B*C-1
  const int t  = threadIdx.x;
  float sm = 0.f, sq = 0.f;
#pragma unroll
  for (int k = 0; k < BLKS_PER_CH / 256; ++k) {   // 2304/256 = 9, fixed order
    sm += psum[(size_t)bc * BLKS_PER_CH + k * 256 + t];
    sq += psq [(size_t)bc * BLKS_PER_CH + k * 256 + t];
  }
  __shared__ float rs[256], rq[256];
  rs[t] = sm; rq[t] = sq;
  __syncthreads();
  for (int o = 128; o > 0; o >>= 1) {
    if (t < o) { rs[t] += rs[t + o]; rq[t] += rq[t + o]; }
    __syncthreads();
  }
  if (t == 0) { csum[bc] = rs[0]; csq[bc] = rq[0]; }
}

// ---------------------------------------------------------------- K3c: fold GN stats into per-channel scale/shift
__global__ void k3c_gn(const float* __restrict__ csum, const float* __restrict__ csq,
                       const float* __restrict__ gnw, const float* __restrict__ gnb,
                       float* __restrict__ gnA, float* __restrict__ gnB) {
  int t = threadIdx.x;
  if (t >= BDIM * CDIM) return;
  int c = t & 63, b = t >> 6;
  int c0 = b * CDIM + ((c >> 1) << 1);      // first channel of this group
  const float n = 2.f * LDIM * LDIM;        // elems per (b, group): 2 channels * L*L
  float mean = (csum[c0] + csum[c0 + 1]) / n;
  float var  = (csq[c0] + csq[c0 + 1]) / n - mean * mean;
  float inv  = rsqrtf(var + EPSV);
  gnA[t] = gnw[c] * inv;
  gnB[t] = gnb[c] - mean * gnw[c] * inv;
}

// ---------------------------------------------------------------- K4: GN+GELU -> 1x1 conv (WMMA) + bias + residual
__global__ __launch_bounds__(256) void k4_pw(const float* __restrict__ h1,
                                             const float* __restrict__ gnA,
                                             const float* __restrict__ gnB,
                                             const float* __restrict__ pww,
                                             const float* __restrict__ pwb,
                                             float* __restrict__ out) {  // holds xp; updated in place
  __shared__ float sG[8][CDIM][17];         // per-wave activations [c_in][j] (padded)
  const int tid  = threadIdx.x;
  const int lane = tid & 31;
  const int w    = tid >> 5;
  const int jt = blockIdx.x, it = blockIdx.y, b = blockIdx.z;

  // A = pw_w  (A[m=c_out_local][k=c_in]), hoisted
  v16bf Aw[4][2];
#pragma unroll
  for (int ct = 0; ct < 4; ++ct)
#pragma unroll
    for (int kh = 0; kh < 2; ++kh)
#pragma unroll
      for (int e = 0; e < 16; ++e) {
        int m = lane & 15;
        int k = wk16(lane, e) + 32 * kh;
        Aw[ct][kh][e] = (__bf16)pww[(ct * 16 + m) * CDIM + k];
      }

  for (int s = 0; s < 2; ++s) {
    const int i = it * 16 + w * 2 + s;

    // load h1 row, apply GN + exact GELU, stage in LDS (coalesced over j)
#pragma unroll 4
    for (int t = 0; t < 32; ++t) {
      int c  = 2 * t + (lane >> 4);
      int jl = lane & 15;
      size_t idx = (((size_t)b * CDIM + c) * LDIM + i) * LDIM + (size_t)(jt * 16 + jl);
      float y = gnA[b * CDIM + c] * h1[idx] + gnB[b * CDIM + c];
      sG[w][c][jl] = 0.5f * y * (1.f + erff(y * 0.70710678118654752f));
    }

    // B[k=c_in][n=j] from LDS
    v16bf Bg[2];
#pragma unroll
    for (int kh = 0; kh < 2; ++kh)
#pragma unroll
      for (int e = 0; e < 16; ++e) {
        int n = lane & 15;
        int k = wk16(lane, e) + 32 * kh;
        Bg[kh][e] = (__bf16)sG[w][k][n];
      }

    v8f acc[4];
#pragma unroll
    for (int ct = 0; ct < 4; ++ct) {
      v8f z{};
      z = __builtin_amdgcn_wmma_f32_16x16x32_bf16(false, Aw[ct][0], false, Bg[0],
                                                  (short)0, z, false, false);
      z = __builtin_amdgcn_wmma_f32_16x16x32_bf16(false, Aw[ct][1], false, Bg[1],
                                                  (short)0, z, false, false);
      acc[ct] = z;
    }

    // out = xp + pw_b + conv ; D: M(=c_out_local)=v+8*(lane>=16), N(=j)=lane&15
#pragma unroll
    for (int ct = 0; ct < 4; ++ct)
#pragma unroll
      for (int v = 0; v < 8; ++v) {
        int c = ct * 16 + v + ((lane >> 4) << 3);
        int j = jt * 16 + (lane & 15);
        size_t idx = (((size_t)b * CDIM + c) * LDIM + i) * LDIM + (size_t)j;
        out[idx] = out[idx] + pwb[c] + acc[ct][v];
      }
  }
}

// ----------------------------------------------------------------
extern "C" void kernel_launch(void* const* d_in, const int* in_sizes, int n_in,
                              void* d_out, int out_size, void* d_ws, size_t ws_size,
                              hipStream_t stream) {
  const float* x_pairwise = (const float*)d_in[0];
  const float* x    = (const float*)d_in[1];
  const float* Wsl  = (const float*)d_in[2];
  const float* bsl  = (const float*)d_in[3];
  const float* Wop  = (const float*)d_in[4];
  const float* bop  = (const float*)d_in[5];
  const float* lnw  = (const float*)d_in[6];
  const float* lnb  = (const float*)d_in[7];
  const float* dww  = (const float*)d_in[8];
  const float* gnw  = (const float*)d_in[9];
  const float* gnb  = (const float*)d_in[10];
  const float* pww  = (const float*)d_in[11];
  const float* pwb  = (const float*)d_in[12];
  float* out = (float*)d_out;

  const size_t nPix   = (size_t)BDIM * CDIM * LDIM * LDIM;   // 75,497,472
  const size_t nBlk   = nPix / 256;                          // 294,912 dwconv blocks
  char* ws = (char*)d_ws;
  size_t off = 0;
  float* h1   = (float*)(ws + off); off += nPix * sizeof(float);
  float* a    = (float*)(ws + off); off += (size_t)BDIM * LDIM * LOWD * sizeof(float);
  float* psum = (float*)(ws + off); off += nBlk * sizeof(float);
  float* psq  = (float*)(ws + off); off += nBlk * sizeof(float);
  float* csum = (float*)(ws + off); off += BDIM * CDIM * sizeof(float);
  float* csq  = (float*)(ws + off); off += BDIM * CDIM * sizeof(float);
  float* gnA  = (float*)(ws + off); off += BDIM * CDIM * sizeof(float);
  float* gnB  = (float*)(ws + off); off += BDIM * CDIM * sizeof(float);

  // a projection
  k1_seq_proj<<<(BDIM * LDIM * LOWD + 255) / 256, 256, 0, stream>>>(x, Wsl, bsl, a);
  // pair block: outer GEMM (WMMA) + LN + residual -> xp in d_out
  k2_pair<<<dim3(LDIM / 16, LDIM / 16, BDIM), 256, 0, stream>>>(
      x_pairwise, a, Wop, bop, lnw, lnb, out);
  // depthwise conv + deterministic GN partial sums
  k3_dwconv<<<(unsigned)nBlk, 256, 0, stream>>>(out, dww, h1, psum, psq);
  // fixed-tree per-channel reduction, then fold into per-channel scale/shift
  k3b_reduce<<<BDIM * CDIM, 256, 0, stream>>>(psum, psq, csum, csq);
  k3c_gn<<<1, 128, 0, stream>>>(csum, csq, gnw, gnb, gnA, gnB);
  // GN + GELU + 1x1 conv (WMMA) + bias + residual, in-place on d_out
  k4_pw<<<dim3(LDIM / 16, LDIM / 16, BDIM), 256, 0, stream>>>(
      h1, gnA, gnB, pww, pwb, out);
}
